// LTFMModel_41266045780230
// MI455X (gfx1250) — matrix operand order repaired and checked by reading, not compiled, not run on hardware
//
#include <hip/hip_runtime.h>
#include <hip/hip_bf16.h>

// ---------------------------------------------------------------------------
// MI455X (gfx1250) fused dual-branch transformer block.
//  * all dense math: v_wmma_f32_16x16x32_f16 (f16 operands, fp32 accumulate)
//  * activations/weights converted to f16 once (halves HBM traffic)
//  * GEMM tile staging: Tensor Data Mover (tensor_load_to_lds, TENSORcnt),
//    double-buffered, with D# LDS padding giving the 80B conflict-free rows
//  * attention K/V staging: global_load_async_to_lds_b128 (ASYNCcnt),
//    double-buffered via in-order ASYNCcnt (wait cnt<=16)
//  * flash attention: score tensor never touches HBM
// ---------------------------------------------------------------------------

typedef __attribute__((ext_vector_type(16))) _Float16 v16h;
typedef __attribute__((ext_vector_type(8)))  _Float16 v8h;
typedef __attribute__((ext_vector_type(8)))  float    v8f;
typedef unsigned int v4u __attribute__((ext_vector_type(4)));
typedef int          v4i __attribute__((ext_vector_type(4)));
typedef int          v8i __attribute__((ext_vector_type(8)));

#define E_DIM   512
#define HID_DIM 2048
#define S_LEN   1024
#define B_SZ    8
#define NH      8
#define HD_DIM  64
#define MROWS   (B_SZ * S_LEN)   // 8192
#define LDA     40               // LDS row stride (f16): 80B, 16B-aligned, conflict-free
#define LDK     72               // K/V tile row stride (f16): 144B, 16B-aligned
#define AW      2                // waves per attention block

__device__ __forceinline__ v8f wmma_f16(v16h a, v16h b, v8f c) {
  return __builtin_amdgcn_wmma_f32_16x16x32_f16(
      false, a, false, b, (short)0, c, false, false);
}

__device__ __forceinline__ unsigned pack2h(_Float16 x, _Float16 y) {
  union { _Float16 h[2]; unsigned u; } p;
  p.h[0] = x; p.h[1] = y;
  return p.u;
}

// CDNA5 async copy: 16B/lane global -> LDS, tracked by ASYNCcnt.
__device__ __forceinline__ void async_copy_b128(unsigned lds_byte_off,
                                                const void* gptr) {
  asm volatile("global_load_async_to_lds_b128 %0, %1, off"
               :: "v"(lds_byte_off), "v"(gptr) : "memory");
}
__device__ __forceinline__ void wait_async0() {
  asm volatile("s_wait_asynccnt 0x0" ::: "memory");
}
// Async loads complete in order: cnt<=16 => the oldest 16 (current tile) landed.
__device__ __forceinline__ void wait_async16() {
  asm volatile("s_wait_asynccnt 0x10" ::: "memory");
}

// ---------------------------------------------------------------------------
// Tensor Data Mover: one descriptor moves a whole (128 x 32) f16 tile
// global->LDS, inserting 16B of LDS padding after every 64B row (-> LDA=40).
// ---------------------------------------------------------------------------
__device__ __forceinline__ v8i tdm_make_g1(int Kstride) {
  v8i g1;
  // [15:0] wg_mask=0 | [17:16] data_size=1 (2B) | [20] pad_en
  // [24:22] pad_interval=3 (16 DW = 64B) | [31:25] pad_amount=3 (4 DW = 16B)
  g1[0] = (int)((1u << 16) | (1u << 20) | (3u << 22) | (3u << 25));
  g1[1] = (int)(32u << 16);    // tensor_dim0 = 32 (elems/row, rel. to tile base)
  g1[2] = (int)(128u << 16);   // tensor_dim0 hi=0 | tensor_dim1 = 128 rows
  g1[3] = (int)(32u << 16);    // tensor_dim1 hi=0 | tile_dim0 = 32
  g1[4] = 128;                 // tile_dim1 = 128 | tile_dim2 = 0
  g1[5] = Kstride;             // tensor_dim0_stride (elems)
  g1[6] = 0;
  g1[7] = 0;
  return g1;
}

__device__ __forceinline__ void tdm_load_tile(unsigned lds_addr,
                                              const void* gaddr, v8i g1) {
  unsigned long long ga = (unsigned long long)gaddr;
  v4u g0;
  g0.x = 1u;                                   // count=1, user descriptor
  g0.y = lds_addr;                             // LDS byte address
  g0.z = (unsigned)ga;                         // global_addr[31:0]
  g0.w = (unsigned)((ga >> 32) & 0x1FFFFFFu) | (2u << 30);  // [56:32] | type=2
  v4i z4 = {0, 0, 0, 0};
#if __clang_major__ >= 23
  v8i z8 = {};
  __builtin_amdgcn_tensor_load_to_lds(g0, g1, z4, z4, z8, 0);
#else
  __builtin_amdgcn_tensor_load_to_lds(g0, g1, z4, z4, 0);
#endif
}

// A-matrix 16x32 f16 fragment from row-major LDS tile (contiguous order: element s holds k = 16*(lane/16) + s).
__device__ __forceinline__ v16h load_frag_a(const _Float16* tile, int lane, int ldk) {
  const int hk = (lane >> 4) ? 16 : 0;
  const _Float16* row = tile + (lane & 15) * ldk;
  v16h a;
  unsigned* ap = reinterpret_cast<unsigned*>(&a);
#pragma unroll
  for (int v = 0; v < 8; ++v) {
    int kk = hk + 2 * v;
    ap[v] = *reinterpret_cast<const unsigned*>(row + kk);
  }
  return a;
}

// B-matrix 32x16 fragment from LDS holding B^T rows (tile[n][k]).
__device__ __forceinline__ v16h load_frag_bt(const _Float16* tile, int lane, int ldk) {
  const int hk = (lane >> 4) ? 16 : 0;
  const _Float16* row = tile + (lane & 15) * ldk + hk;
  v16h b;
  unsigned* bp = reinterpret_cast<unsigned*>(&b);
#pragma unroll
  for (int v = 0; v < 8; ++v)
    bp[v] = *reinterpret_cast<const unsigned*>(row + 2 * v);
  return b;
}

// ---------------------------------------------------------------------------
// GEMM: C[M,N](f16) = epi(A[M,K](f16) @ Wt[N,K]^T(f16) + bias (+Cin)).
// 128x128x32 tiles; TDM double-buffered staging; 8 waves x (4x2 WMMA).
// Epilogue specialized at compile time: straight-line stores, no branches.
// ---------------------------------------------------------------------------
template <bool GELU, bool ACC>
__global__ __launch_bounds__(256) void gemm_kernel(
    const _Float16* __restrict__ A, const _Float16* __restrict__ Wt,
    const float* __restrict__ bias, const _Float16* __restrict__ Cin,
    _Float16* __restrict__ C, int N, int K) {
  __shared__ __align__(16) _Float16 sA[2][128 * LDA];
  __shared__ __align__(16) _Float16 sB[2][128 * LDA];
  const int tid  = threadIdx.x;
  const int lane = tid & 31, wave = tid >> 5;
  const int wm = wave & 1, wn = wave >> 1;           // 2 x 4 wave grid
  const int m0 = blockIdx.y * 128, n0 = blockIdx.x * 128;
  const unsigned baA[2] = {(unsigned)(unsigned long long)&sA[0][0],
                           (unsigned)(unsigned long long)&sA[1][0]};
  const unsigned baB[2] = {(unsigned)(unsigned long long)&sB[0][0],
                           (unsigned)(unsigned long long)&sB[1][0]};
  const v8i g1 = tdm_make_g1(K);

  auto stage = [&](int buf, int k0) {                // one wave drives the TDM
    if (wave == 0) {
      tdm_load_tile(baA[buf], A + (size_t)m0 * K + k0, g1);
      tdm_load_tile(baB[buf], Wt + (size_t)n0 * K + k0, g1);
    }
  };

  const v8f vzero = {};
  v8f acc[4][2];
#pragma unroll
  for (int i = 0; i < 4; ++i)
#pragma unroll
    for (int j = 0; j < 2; ++j) acc[i][j] = vzero;

  stage(0, 0);
  if (wave == 0) __builtin_amdgcn_s_wait_tensorcnt(0);
  __syncthreads();

  int buf = 0;
  for (int k0 = 0; k0 < K; k0 += 32) {
    if (k0 + 32 < K) stage(buf ^ 1, k0 + 32);       // DMA overlaps WMMAs

    v16h af[4], bf[2];
#pragma unroll
    for (int mt = 0; mt < 4; ++mt)
      af[mt] = load_frag_a(&sA[buf][(wm * 64 + mt * 16) * LDA], lane, LDA);
#pragma unroll
    for (int nt = 0; nt < 2; ++nt)
      bf[nt] = load_frag_bt(&sB[buf][(wn * 32 + nt * 16) * LDA], lane, LDA);
#pragma unroll
    for (int mt = 0; mt < 4; ++mt)
#pragma unroll
      for (int nt = 0; nt < 2; ++nt)
        acc[mt][nt] = wmma_f16(af[mt], bf[nt], acc[mt][nt]);

    if (wave == 0) __builtin_amdgcn_s_wait_tensorcnt(0);  // next buf landed
    __syncthreads();
    buf ^= 1;
  }

  const int hm = (lane >> 4) ? 8 : 0;
#pragma unroll
  for (int mt = 0; mt < 4; ++mt) {
#pragma unroll
    for (int nt = 0; nt < 2; ++nt) {
      int col = n0 + wn * 32 + nt * 16 + (lane & 15);
      float bv = bias ? bias[col] : 0.f;
#pragma unroll
      for (int r = 0; r < 8; ++r) {
        int row = m0 + wm * 64 + mt * 16 + hm + r;
        float x = acc[mt][nt][r] + bv;
        if (ACC)  x += (float)Cin[(size_t)row * N + col];
        if (GELU) x = 0.5f * x * (1.f + erff(x * 0.70710678118f));
        C[(size_t)row * N + col] = (_Float16)x;
      }
    }
  }
}

// ---------------------------------------------------------------------------
// Flash attention, f16 in/out. AW waves/block; one wave per (b,h,16-row qtile).
// K/V 32x64 head tiles double-buffered per wave via async b128 copies:
// issue next tile's 16 copies, then s_wait_asynccnt 16 (in-order completion
// guarantees the current tile's copies - the oldest 16 - have landed).
// ---------------------------------------------------------------------------
__global__ __launch_bounds__(32 * AW) void attn_kernel(
    const _Float16* __restrict__ Qb, const _Float16* __restrict__ Kb,
    const _Float16* __restrict__ Vb, _Float16* __restrict__ Ob) {
  __shared__ __align__(16) _Float16 sK[AW][2][32 * LDK];
  __shared__ __align__(16) _Float16 sV[AW][2][32 * LDK];
  __shared__ __align__(16) _Float16 sP[AW][16 * LDA];
  const int lane = threadIdx.x & 31, wave = threadIdx.x >> 5;
  const int w  = blockIdx.x * AW + wave;
  const int qt = w & 63;
  const int h  = (w >> 6) & 7;
  const int b  = w >> 9;
  const int hm   = (lane >> 4) ? 8 : 0;
  const int hb   = (lane >> 4) ? 16 : 0;
  const int lmod = lane & 15;
  const size_t base = ((size_t)b * S_LEN) * E_DIM + (size_t)h * HD_DIM;
  const _Float16* Q = Qb + base;
  const _Float16* K = Kb + base;
  const _Float16* V = Vb + base;
  _Float16* O = Ob + base;
  _Float16* myP = &sP[wave][0];
  const unsigned kb_base[2] = {(unsigned)(unsigned long long)&sK[wave][0][0],
                               (unsigned)(unsigned long long)&sK[wave][1][0]};
  const unsigned vb_base[2] = {(unsigned)(unsigned long long)&sV[wave][0][0],
                               (unsigned)(unsigned long long)&sV[wave][1][0]};

  auto issue = [&](int buf, int kt) {   // 16 async b128 copies (8 K + 8 V)
    const int k0 = kt * 32;
#pragma unroll
    for (int i = 0; i < 8; ++i) {
      int lin = lane + i * 32;                     // 256 x 16B chunks
      int r = lin >> 3, c = (lin & 7) * 8;
      unsigned off = (unsigned)(r * LDK + c) * 2u;
      async_copy_b128(kb_base[buf] + off, K + (size_t)(k0 + r) * E_DIM + c);
      async_copy_b128(vb_base[buf] + off, V + (size_t)(k0 + r) * E_DIM + c);
    }
  };

  v16h qf[2];
  {
    const _Float16* qr = Q + (size_t)(qt * 16 + lmod) * E_DIM;
#pragma unroll
    for (int f = 0; f < 2; ++f) {
      unsigned* ap = reinterpret_cast<unsigned*>(&qf[f]);
#pragma unroll
      for (int v = 0; v < 8; ++v) {
        int kk = 32 * f + hb + 2 * v;
        ap[v] = *reinterpret_cast<const unsigned*>(qr + kk);
      }
    }
  }

  const v8f vzero = {};
  float mrow[8], ssum[8];
  v8f o[4];
#pragma unroll
  for (int r = 0; r < 8; ++r) { mrow[r] = -3.0e38f; ssum[r] = 0.f; }
#pragma unroll
  for (int t = 0; t < 4; ++t) o[t] = vzero;

  issue(0, 0);
  int buf = 0;
  for (int kt = 0; kt < S_LEN / 32; ++kt) {
    const bool more = (kt + 1) < (S_LEN / 32);
    if (more) issue(buf ^ 1, kt + 1);   // prefetch next tile into other buffer
    if (more) wait_async16(); else wait_async0();
    const _Float16* sKp = &sK[wave][buf][0];
    const _Float16* sVp = &sV[wave][buf][0];

    v8f s0 = vzero, s1 = vzero;
#pragma unroll
    for (int f = 0; f < 2; ++f) {
      v16h kb0, kb1;
      unsigned* p0 = reinterpret_cast<unsigned*>(&kb0);
      unsigned* p1 = reinterpret_cast<unsigned*>(&kb1);
      const _Float16* kr0 = sKp + lmod * LDK + 32 * f + hb;
      const _Float16* kr1 = sKp + (16 + lmod) * LDK + 32 * f + hb;
#pragma unroll
      for (int v = 0; v < 8; ++v) {
        p0[v] = *reinterpret_cast<const unsigned*>(kr0 + 2 * v);
        p1[v] = *reinterpret_cast<const unsigned*>(kr1 + 2 * v);
      }
      s0 = wmma_f16(qf[f], kb0, s0);
      s1 = wmma_f16(qf[f], kb1, s1);
    }
    float fs[8];
#pragma unroll
    for (int r = 0; r < 8; ++r) {
      float a0 = s0[r] * 0.125f, a1 = s1[r] * 0.125f;
      float vmax = fmaxf(a0, a1);
#pragma unroll
      for (int m = 1; m <= 8; m <<= 1) vmax = fmaxf(vmax, __shfl_xor(vmax, m, 32));
      float mn  = fmaxf(mrow[r], vmax);
      float p0v = __expf(a0 - mn);
      float p1v = __expf(a1 - mn);
      float ps  = p0v + p1v;
#pragma unroll
      for (int m = 1; m <= 8; m <<= 1) ps += __shfl_xor(ps, m, 32);
      float f = __expf(mrow[r] - mn);
      ssum[r] = ssum[r] * f + ps;
      mrow[r] = mn;
      fs[r] = f;
      myP[(r + hm) * LDA + lmod]      = (_Float16)p0v;
      myP[(r + hm) * LDA + 16 + lmod] = (_Float16)p1v;
    }
#pragma unroll
    for (int t = 0; t < 4; ++t)
#pragma unroll
      for (int r = 0; r < 8; ++r) o[t][r] *= fs[r];

    v16h pf = load_frag_a(myP, lane, LDA);
#pragma unroll
    for (int t = 0; t < 4; ++t) {
      v16h vf;
      unsigned* bp = reinterpret_cast<unsigned*>(&vf);
      const _Float16* vc = sVp + t * 16 + lmod;
#pragma unroll
      for (int v = 0; v < 8; ++v)
        bp[v] = pack2h(vc[(hb + 2 * v) * LDK], vc[(hb + 2 * v + 1) * LDK]);
      o[t] = wmma_f16(pf, vf, o[t]);
    }
    buf ^= 1;
  }
#pragma unroll
  for (int r = 0; r < 8; ++r) {
    float inv = 1.f / ssum[r];
    int row = qt * 16 + hm + r;
    _Float16* orow = O + (size_t)row * E_DIM;
#pragma unroll
    for (int t = 0; t < 4; ++t)
      orow[t * 16 + lmod] = (_Float16)(o[t][r] * inv);
  }
}

// ---------------------------------------------------------------------------
// LayerNorm over E=512 (f16 in/out, fp32 math), optional fused residual.
// ---------------------------------------------------------------------------
__global__ __launch_bounds__(256) void ln_kernel(
    const _Float16* __restrict__ x, const _Float16* __restrict__ res,
    const float* __restrict__ g, const float* __restrict__ bta,
    _Float16* __restrict__ out) {
  const int lane = threadIdx.x & 31, wave = threadIdx.x >> 5;
  const size_t row = (size_t)blockIdx.x * 8 + wave;
  const _Float16* xr = x + row * E_DIM;
  float vals[16];
  float s = 0.f;
#pragma unroll
  for (int i = 0; i < 2; ++i) {
    int c = (lane + i * 32) * 8;
    v8h hv = *reinterpret_cast<const v8h*>(xr + c);
#pragma unroll
    for (int j = 0; j < 8; ++j) vals[i * 8 + j] = (float)hv[j];
    if (res) {
      v8h rv = *reinterpret_cast<const v8h*>(res + row * E_DIM + c);
#pragma unroll
      for (int j = 0; j < 8; ++j) vals[i * 8 + j] += (float)rv[j];
    }
#pragma unroll
    for (int j = 0; j < 8; ++j) s += vals[i * 8 + j];
  }
#pragma unroll
  for (int m = 1; m <= 16; m <<= 1) s += __shfl_xor(s, m, 32);
  const float mean = s * (1.f / E_DIM);
  float vs = 0.f;
#pragma unroll
  for (int j = 0; j < 16; ++j) { float d = vals[j] - mean; vs += d * d; }
#pragma unroll
  for (int m = 1; m <= 16; m <<= 1) vs += __shfl_xor(vs, m, 32);
  const float rstd = rsqrtf(vs * (1.f / E_DIM) + 1e-5f);
  _Float16* orow = out + row * E_DIM;
#pragma unroll
  for (int i = 0; i < 2; ++i) {
    int c = (lane + i * 32) * 8;
    v8h ov;
#pragma unroll
    for (int j = 0; j < 8; ++j)
      ov[j] = (_Float16)((vals[i * 8 + j] - mean) * rstd * g[c + j] + bta[c + j]);
    *reinterpret_cast<v8h*>(orow + c) = ov;
  }
}

// fp32 -> f16 (activations)
__global__ __launch_bounds__(256) void cvt_kernel(
    const float* __restrict__ in, _Float16* __restrict__ out) {
  size_t i = ((size_t)blockIdx.x * 256 + threadIdx.x) * 4;
  float4 v = *reinterpret_cast<const float4*>(in + i);
  out[i + 0] = (_Float16)v.x; out[i + 1] = (_Float16)v.y;
  out[i + 2] = (_Float16)v.z; out[i + 3] = (_Float16)v.w;
}

// W[K,N] fp32 -> Wt[N,K] f16 (one-time; coalesced writes)
__global__ __launch_bounds__(256) void cvt_wt_kernel(
    const float* __restrict__ W, _Float16* __restrict__ Wt, int K, int N) {
  size_t lin = (size_t)blockIdx.x * 256 + threadIdx.x;  // over N*K
  int n = (int)(lin / K), k = (int)(lin % K);
  Wt[lin] = (_Float16)W[(size_t)k * N + n];
}

// out(fp32) = m(f16) + f(f16)
__global__ __launch_bounds__(256) void add_kernel(
    const _Float16* __restrict__ a, const _Float16* __restrict__ b,
    float* __restrict__ o) {
  size_t i = ((size_t)blockIdx.x * 256 + threadIdx.x) * 4;
#pragma unroll
  for (int j = 0; j < 4; ++j)
    o[i + j] = (float)a[i + j] + (float)b[i + j];
}

// ---------------------------------------------------------------------------
// Orchestration
// ---------------------------------------------------------------------------
extern "C" void kernel_launch(void* const* d_in, const int* in_sizes, int n_in,
                              void* d_out, int out_size, void* d_ws, size_t ws_size,
                              hipStream_t stream) {
  (void)in_sizes; (void)n_in; (void)out_size; (void)ws_size;
  const float* const* P = (const float* const*)d_in;
  const int GA = 2, LA = 10, FM = 18;
  const float* gn1_g = P[26]; const float* gn1_b = P[27];
  const float* gn2_g = P[28]; const float* gn2_b = P[29];
  const float* ln1_g = P[30]; const float* ln1_b = P[31];
  const float* ln2_g = P[32]; const float* ln2_b = P[33];
  const float* mn_g  = P[34]; const float* mn_b  = P[35];
  const float* fn_g  = P[36]; const float* fn_b  = P[37];
  const float* ffn_w1 = P[38]; const float* ffn_b1 = P[39];
  const float* ffn_w2 = P[40]; const float* ffn_b2 = P[41];
  const float* fus_w  = P[42]; const float* fus_b  = P[43];

  // Workspace layout (f16 elements). ~150 MB total.
  const size_t U  = (size_t)MROWS * E_DIM;
  const size_t EE = (size_t)E_DIM * E_DIM;
  _Float16* ws   = (_Float16*)d_ws;
  _Float16* xg16 = ws;
  _Float16* xl16 = xg16 + U;
  _Float16* q    = xl16 + U;
  _Float16* kbuf = q + U;
  _Float16* vbuf = kbuf + U;
  _Float16* att  = vbuf + U;
  _Float16* tmp  = att + U;
  _Float16* g1   = tmp + U;
  _Float16* g2   = g1 + U;
  _Float16* l1   = g2 + U;
  _Float16* l2   = l1 + U;
  _Float16* mbuf = l2 + U;
  _Float16* fbuf = mbuf + U;
  _Float16* fpre = fbuf + U;
  _Float16* hid  = fpre + U;
  _Float16* wt   = hid + (size_t)MROWS * HID_DIM;
  _Float16* wqkvo[12];
  for (int i = 0; i < 12; ++i) wqkvo[i] = wt + (size_t)i * EE;
  _Float16* w_ffn1 = wt + 12 * EE;                          // [HID][E]
  _Float16* w_ffn2 = w_ffn1 + (size_t)HID_DIM * E_DIM;      // [E][HID]
  _Float16* w_fus0 = w_ffn2 + (size_t)E_DIM * HID_DIM;      // [E][E]
  _Float16* w_fus1 = w_fus0 + EE;                           // [E][E]

  cvt_kernel<<<(int)(U / 1024), 256, 0, stream>>>((const float*)d_in[0], xg16);
  cvt_kernel<<<(int)(U / 1024), 256, 0, stream>>>((const float*)d_in[1], xl16);
  for (int s = 0; s < 3; ++s)
    for (int j = 0; j < 4; ++j)
      cvt_wt_kernel<<<(int)(EE / 256), 256, 0, stream>>>(
          P[2 + s * 8 + j * 2], wqkvo[s * 4 + j], E_DIM, E_DIM);
  cvt_wt_kernel<<<(int)((size_t)E_DIM * HID_DIM / 256), 256, 0, stream>>>(
      ffn_w1, w_ffn1, E_DIM, HID_DIM);
  cvt_wt_kernel<<<(int)((size_t)E_DIM * HID_DIM / 256), 256, 0, stream>>>(
      ffn_w2, w_ffn2, HID_DIM, E_DIM);
  cvt_wt_kernel<<<(int)(EE / 256), 256, 0, stream>>>(fus_w, w_fus0, E_DIM, E_DIM);
  cvt_wt_kernel<<<(int)(EE / 256), 256, 0, stream>>>(fus_w + EE, w_fus1, E_DIM, E_DIM);

  auto gemm = [&](const _Float16* A, const _Float16* Wt, const float* bias,
                  _Float16* C, int N, int K) {
    dim3 grid(N / 128, MROWS / 128);
    gemm_kernel<false, false><<<grid, 256, 0, stream>>>(A, Wt, bias, nullptr, C, N, K);
  };
  auto gemm_gelu = [&](const _Float16* A, const _Float16* Wt, const float* bias,
                       _Float16* C, int N, int K) {
    dim3 grid(N / 128, MROWS / 128);
    gemm_kernel<true, false><<<grid, 256, 0, stream>>>(A, Wt, bias, nullptr, C, N, K);
  };
  auto gemm_acc = [&](const _Float16* A, const _Float16* Wt,
                      const _Float16* Cin, _Float16* C, int N, int K) {
    dim3 grid(N / 128, MROWS / 128);
    gemm_kernel<false, true><<<grid, 256, 0, stream>>>(A, Wt, nullptr, Cin, C, N, K);
  };
  auto attn = [&](const _Float16* Q, const _Float16* K, const _Float16* V,
                  _Float16* O) {
    attn_kernel<<<(B_SZ * NH * (S_LEN / 16)) / AW, 32 * AW, 0, stream>>>(Q, K, V, O);
  };
  auto ln = [&](const _Float16* x, const _Float16* res, const float* g,
                const float* b, _Float16* o) {
    ln_kernel<<<MROWS / 8, 256, 0, stream>>>(x, res, g, b, o);
  };

  auto branch = [&](const _Float16* X, int AP, int WSET,
                    const float* n1g, const float* n1b,
                    const float* n2g, const float* n2b,
                    _Float16* s1, _Float16* s2) {
    gemm(X, wqkvo[WSET + 0], P[AP + 1], q,    E_DIM, E_DIM);
    gemm(X, wqkvo[WSET + 1], P[AP + 3], kbuf, E_DIM, E_DIM);
    gemm(X, wqkvo[WSET + 2], P[AP + 5], vbuf, E_DIM, E_DIM);
    attn(q, kbuf, vbuf, att);
    gemm(att, wqkvo[WSET + 3], P[AP + 7], tmp, E_DIM, E_DIM);
    ln(X, tmp, n1g, n1b, s1);
    gemm_gelu(s1, w_ffn1, ffn_b1, hid, HID_DIM, E_DIM);
    gemm(hid, w_ffn2, ffn_b2, tmp, E_DIM, HID_DIM);
    ln(s1, tmp, n2g, n2b, s2);
  };

  branch(xg16, GA, 0, gn1_g, gn1_b, gn2_g, gn2_b, g1, g2);
  branch(xl16, LA, 4, ln1_g, ln1_b, ln2_g, ln2_b, l1, l2);

  gemm(g2, wqkvo[8],  P[FM + 1], q,    E_DIM, E_DIM);
  gemm(l2, wqkvo[9],  P[FM + 3], kbuf, E_DIM, E_DIM);
  gemm(l2, wqkvo[10], P[FM + 5], vbuf, E_DIM, E_DIM);
  attn(q, kbuf, vbuf, att);
  gemm(att, wqkvo[11], P[FM + 7], tmp, E_DIM, E_DIM);
  ln(g2, tmp, mn_g, mn_b, mbuf);

  gemm(g2, w_fus0, fus_b, fpre, E_DIM, E_DIM);
  gemm_acc(l2, w_fus1, fpre, fpre, E_DIM, E_DIM);
  ln(fpre, nullptr, fn_g, fn_b, fbuf);

  add_kernel<<<(int)(U / 1024), 256, 0, stream>>>(mbuf, fbuf, (float*)d_out);
}
